// GATNetwork_24180665876525
// MI455X (gfx1250) — compile-verified, matrix operand order checked by reading, not run on hardware
//
#include <hip/hip_runtime.h>
#include <hip/hip_bf16.h>

// ---------------------------------------------------------------------------
// GAT network for MI455X (gfx1250, wave32, WMMA + Tensor Data Mover).
// GEMMs: bf16 WMMA (v_wmma_f32_16x16x32_bf16); weight panel pre-packed into
// k-pair dwords (WMMA B operand layout), DMA'd to LDS once per block by
// tensor_load_to_lds (TENSORcnt). Graph passes use flat float atomics
// (L2-resident: feature matrix is 20MB << 192MB L2).
// ---------------------------------------------------------------------------

typedef __attribute__((ext_vector_type(16))) __bf16        v16bf;
typedef __attribute__((ext_vector_type(8)))  float         v8f;
typedef __attribute__((ext_vector_type(4)))  unsigned int  u32x4;
typedef __attribute__((ext_vector_type(8)))  int           i32x8;
typedef __attribute__((ext_vector_type(4)))  int           i32x4;

#define NN  20000
#define EE  640000
#define ETOT 660000   // EE + NN self loops
#define GG  64

#define TM 128        // rows per block (2 x 16-row subtiles per wave)
#define TN 64
#define TK 32
#define KMAX 256      // max K of any layer

// ------------------------------ helpers ------------------------------------

__device__ inline unsigned short f32_to_bf16_bits(float x) {
    unsigned int u = __float_as_uint(x);
    u += 0x7FFFu + ((u >> 16) & 1u);      // round-to-nearest-even
    return (unsigned short)(u >> 16);
}

__global__ void f32_to_bf16_k(const float* __restrict__ in,
                              unsigned short* __restrict__ out, size_t n) {
    size_t i = (size_t)blockIdx.x * blockDim.x + threadIdx.x;
    if (i < n) out[i] = f32_to_bf16_bits(in[i]);
}

// pack W[k][n],W[k+1][n] -> Wp[k/2][n] dword (WMMA B-operand pair layout)
__global__ void pack_w_k(const float* __restrict__ W,
                         unsigned int* __restrict__ Wp, int K2, int N) {
    int i = blockIdx.x * blockDim.x + threadIdx.x;
    if (i >= K2 * N) return;
    int kp = i / N, n = i % N;
    unsigned int lo = f32_to_bf16_bits(W[(size_t)(2 * kp)     * N + n]);
    unsigned int hi = f32_to_bf16_bits(W[(size_t)(2 * kp + 1) * N + n]);
    Wp[i] = lo | (hi << 16);
}

__global__ void fill_k(float* __restrict__ p, float v, size_t n) {
    size_t i = (size_t)blockIdx.x * blockDim.x + threadIdx.x;
    if (i < n) p[i] = v;
}

// monotonic float atomic max (int/uint ordering trick)
__device__ inline void atomicMaxF(float* addr, float v) {
    if (v >= 0.0f) atomicMax((int*)addr, __float_as_int(v));
    else           atomicMin((unsigned int*)addr, __float_as_uint(v));
}

__device__ inline void edge_nodes(const long long* __restrict__ ei, int e,
                                  int& s, int& d) {
    if (e < EE) { s = (int)ei[e]; d = (int)ei[EE + e]; }
    else        { s = d = e - EE; }               // self loop
}

// ------------------------------ WMMA GEMM ----------------------------------
// C[M,N] = A[M,K] x B[K,N]; A bf16 row-major, Bp k-pair-packed dwords
// [K/2][N], C f32 row-major. Block: 128 threads = 4 waves; 128x64 C tile.
// Each wave owns 2 M-subtiles so every B fragment feeds 2 WMMAs.
// Bp panel [K/2 x 64] dwords DMA'd to LDS once by the Tensor Data Mover.
__global__ __launch_bounds__(128)
void wmma_gemm_bf16(const unsigned short* __restrict__ A,
                    const unsigned int* __restrict__ Bp,
                    float* __restrict__ C, int M, int K, int N) {
    __shared__ unsigned short As[TM * TK];            // [row][k]          8 KB
    __shared__ unsigned int   Bs[(KMAX / 2) * TN];    // [kpair][n] panel 32 KB

    const int tid  = threadIdx.x;
    const int wave = tid >> 5;
    const int lane = tid & 31;
    const int m0 = blockIdx.x * TM;
    const int n0 = blockIdx.y * TN;
    const int K2 = K >> 1;

    // ---- TDM: DMA the Bp panel [K/2][64] dwords (row stride N) to LDS ----
    if (tid < 32) {  // wave 0 only; TDM ignores EXEC, one DMA per block
        unsigned long long ga = (unsigned long long)Bp + (unsigned long long)n0 * 4ull;
        unsigned int lds0 = (unsigned int)(size_t)(&Bs[0]);   // LDS offset = addr[31:0]
        u32x4 g0;
        g0.x = 1u;                                            // count=1, user mode
        g0.y = lds0;                                          // lds_addr
        g0.z = (unsigned int)(ga & 0xFFFFFFFFull);            // global_addr[31:0]
        g0.w = (unsigned int)((ga >> 32) & 0x1FFFFFFull)      // global_addr[56:32]
             | (2u << 30);                                    // type=2 ("image")
        i32x8 g1;
        g1[0] = 0x00020000;                                   // data_size=2 (4 bytes)
        g1[1] = (int)(64u << 16);                             // tensor_dim0 = 64
        g1[2] = (int)(((unsigned int)K2 & 0xFFFFu) << 16);    // tensor_dim1 = K/2
        g1[3] = (int)(64u << 16);                             // tile_dim0 = 64
        g1[4] = K2;                                           // tile_dim1 = K/2
        g1[5] = N;                                            // tensor_dim0_stride = N
        g1[6] = 0;
        g1[7] = 0;
        i32x4 gz4 = {0, 0, 0, 0};                             // 2-D tensor: groups 2/3 unused
        i32x8 gz8 = {0, 0, 0, 0, 0, 0, 0, 0};
        __builtin_amdgcn_tensor_load_to_lds(g0, g1, gz4, gz4, gz8, 0);
        __builtin_amdgcn_s_wait_tensorcnt((short)0);
    }
    __syncthreads();                                          // publish B panel

    v8f acc[2][4] = {};

    const int r     = lane & 15;         // row (A) / col (B) within 16
    const int koff  = (lane >> 4) * 8;   // K half selector per WMMA layout
    const int kpoff = koff >> 1;         // pair-index offset (0 or 4)

    for (int k0 = 0; k0 < K; k0 += TK) {
        // ---- cooperative load A tile (128x32 bf16 = 8KB) via b128 loads ----
#pragma unroll
        for (int it = 0; it < 4; ++it) {
            int idx = tid + it * 128;        // 512 uint4 total
            int row = idx >> 2;              // 4 uint4 per 32-elem row
            int seg = idx & 3;
            int gr  = m0 + row;
            uint4 v;
            if (gr < M) v = *(const uint4*)(A + (size_t)gr * K + k0 + seg * 8);
            else        v = make_uint4(0u, 0u, 0u, 0u);
            *(uint4*)(&As[row * TK + seg * 8]) = v;
        }
        // prefetch next A tile (global_prefetch_b8)
        if (k0 + TK < K) {
            int pr = m0 + tid;
            if (pr >= M) pr = M - 1;
            __builtin_prefetch(A + (size_t)pr * K + k0 + TK, 0, 0);
        }
        __syncthreads();

        // ---- build 2 A fragments: subtile s rows = s*64 + 16w + r ----
        union { unsigned int u[8]; v16bf v; } af[2];
#pragma unroll
        for (int s = 0; s < 2; ++s) {
            const unsigned short* arow = &As[(s * 64 + wave * 16 + r) * TK];
#pragma unroll
            for (int j = 0; j < 4; ++j) {
                af[s].u[j]     = *(const unsigned int*)(arow + koff + 2 * j);
                af[s].u[4 + j] = *(const unsigned int*)(arow + koff + 16 + 2 * j);
            }
        }
        // ---- 4 N-tiles: B fragment = 8 x ds_load_b32, feeds both M-subtiles ----
        const int kp0 = (k0 >> 1) + kpoff;
#pragma unroll
        for (int t = 0; t < 4; ++t) {
            union { unsigned int u[8]; v16bf v; } bf;
            const int col = t * 16 + r;
#pragma unroll
            for (int j = 0; j < 4; ++j) {
                bf.u[j]     = Bs[(kp0 + j)     * TN + col];   // k pairs koff+2j,+1
                bf.u[4 + j] = Bs[(kp0 + 8 + j) * TN + col];   // k pairs koff+16+2j,+1
            }
            acc[0][t] = __builtin_amdgcn_wmma_f32_16x16x32_bf16(
                false, af[0].v, false, bf.v, (short)0, acc[0][t], false, false);
            acc[1][t] = __builtin_amdgcn_wmma_f32_16x16x32_bf16(
                false, af[1].v, false, bf.v, (short)0, acc[1][t], false, false);
        }
        __syncthreads();
    }

    // ---- epilogue: C layout = VGPR rr -> row rr (+8 for lanes>=16), col lane&15
    const int rowadd = (lane >> 4) * 8;
#pragma unroll
    for (int s = 0; s < 2; ++s) {
#pragma unroll
        for (int t = 0; t < 4; ++t) {
            int col = n0 + t * 16 + r;
#pragma unroll
            for (int rr = 0; rr < 8; ++rr) {
                int row = m0 + s * 64 + wave * 16 + rr + rowadd;
                if (row < M) C[(size_t)row * N + col] = acc[s][t][rr];
            }
        }
    }
}

// ------------------------- attention scalar dots ---------------------------
__global__ void attn_dots_k(const float* __restrict__ h,
                            const float* __restrict__ a_s,
                            const float* __restrict__ a_d,
                            float* __restrict__ asrc, float* __restrict__ adst,
                            int Hh, int Cc) {
    int i = blockIdx.x * blockDim.x + threadIdx.x;
    if (i >= NN * Hh) return;
    int n = i / Hh, hh = i % Hh;
    const float* hp = h + (size_t)n * Hh * Cc + hh * Cc;
    const float* as = a_s + hh * Cc;
    const float* ad = a_d + hh * Cc;
    float s1 = 0.f, s2 = 0.f;
    for (int c = 0; c < Cc; ++c) { float v = hp[c]; s1 += v * as[c]; s2 += v * ad[c]; }
    asrc[i] = s1; adst[i] = s2;
}

// --------------------------- edge softmax passes ---------------------------

__device__ inline float edge_logit(const float* asrc, const float* adst,
                                   int s, int d, int hh, int Hh) {
    float v = asrc[s * Hh + hh] + adst[d * Hh + hh];
    return v < 0.f ? 0.2f * v : v;        // leaky_relu(0.2)
}

__global__ void edge_max_k(const long long* __restrict__ ei,
                           const float* __restrict__ asrc,
                           const float* __restrict__ adst,
                           float* __restrict__ m, int Hh) {
    int t = blockIdx.x * blockDim.x + threadIdx.x;
    if (t >= ETOT * Hh) return;
    int e = t / Hh, hh = t % Hh;
    int s, d; edge_nodes(ei, e, s, d);
    atomicMaxF(&m[d * Hh + hh], edge_logit(asrc, adst, s, d, hh, Hh));
}

__global__ void edge_exp_k(const long long* __restrict__ ei,
                           const float* __restrict__ asrc,
                           const float* __restrict__ adst,
                           const float* __restrict__ m,
                           float* __restrict__ denom,
                           float* __restrict__ alpha, int Hh) {
    int t = blockIdx.x * blockDim.x + threadIdx.x;
    if (t >= ETOT * Hh) return;
    int e = t / Hh, hh = t % Hh;
    int s, d; edge_nodes(ei, e, s, d);
    float ex = __expf(edge_logit(asrc, adst, s, d, hh, Hh) - m[d * Hh + hh]);
    alpha[t] = ex;
    atomicAdd(&denom[d * Hh + hh], ex);
}

__global__ void edge_norm_k(const long long* __restrict__ ei,
                            const float* __restrict__ denom,
                            float* __restrict__ alpha, int Hh) {
    int t = blockIdx.x * blockDim.x + threadIdx.x;
    if (t >= ETOT * Hh) return;
    int e = t / Hh, hh = t % Hh;
    int s, d; edge_nodes(ei, e, s, d);
    alpha[t] = alpha[t] / (denom[d * Hh + hh] + 1e-16f);
}

// ----------------------- edge-weighted aggregation -------------------------
__global__ void edge_aggr_k(const long long* __restrict__ ei,
                            const float* __restrict__ alpha,
                            const float* __restrict__ h,
                            float* __restrict__ out, int Hh, int Cc) {
    int F = Hh * Cc;
    size_t t = (size_t)blockIdx.x * blockDim.x + threadIdx.x;
    if (t >= (size_t)ETOT * F) return;
    int e = (int)(t / F);
    int f = (int)(t % F);
    int hh = f / Cc;
    int s, d; edge_nodes(ei, e, s, d);
    atomicAdd(&out[(size_t)d * F + f], alpha[e * Hh + hh] * h[(size_t)s * F + f]);
}

// ----------------------------- bias + ELU ----------------------------------
__global__ void bias_act_k(const float* __restrict__ in,
                           const float* __restrict__ b,
                           float* __restrict__ out, int F, int do_elu) {
    size_t i = (size_t)blockIdx.x * blockDim.x + threadIdx.x;
    if (i >= (size_t)NN * F) return;
    int f = (int)(i % F);
    float v = in[i] + b[f];
    if (do_elu) v = v > 0.f ? v : (__expf(v) - 1.f);
    out[i] = v;
}

// ------------------------------- pooling -----------------------------------
__global__ void pool_sum_k(const long long* __restrict__ batch,
                           const float* __restrict__ h,
                           float* __restrict__ psum) {
    int t = blockIdx.x * blockDim.x + threadIdx.x;
    if (t >= NN * 128) return;
    int n = t / 128, f = t % 128;
    int g = (int)batch[n];
    atomicAdd(&psum[g * 128 + f], h[(size_t)n * 128 + f]);
}

__global__ void pool_cnt_k(const long long* __restrict__ batch,
                           float* __restrict__ cnt) {
    int n = blockIdx.x * blockDim.x + threadIdx.x;
    if (n >= NN) return;
    atomicAdd(&cnt[(int)batch[n]], 1.0f);
}

__global__ void pool_fin_k(const float* __restrict__ psum,
                           const float* __restrict__ cnt,
                           float* __restrict__ out) {
    int t = blockIdx.x * blockDim.x + threadIdx.x;
    if (t >= GG * 128) return;
    out[t] = psum[t] / fmaxf(cnt[t / 128], 1.0f);
}

// ------------------------------ host driver --------------------------------

static inline unsigned int nblk(size_t n, int b) { return (unsigned int)((n + b - 1) / b); }

struct Scratch {
    unsigned short *xb;
    unsigned int   *wb;
    float *hlin, *hagg, *hbuf, *asrc, *adst, *mbuf, *dbuf, *alpha, *psum, *pcnt;
};

static void run_layer(const float* xin, int K, const float* W,
                      const float* a_s, const float* a_d, const float* bias,
                      int Hh, int Cc, int do_elu, float* fout,
                      const long long* ei, const Scratch& S, hipStream_t stream) {
    const int Nout = Hh * Cc;
    size_t nx = (size_t)NN * K;
    f32_to_bf16_k<<<nblk(nx, 256), 256, 0, stream>>>(xin, S.xb, nx);
    int npk = (K / 2) * Nout;
    pack_w_k<<<nblk((size_t)npk, 256), 256, 0, stream>>>(W, S.wb, K / 2, Nout);

    dim3 gg((NN + TM - 1) / TM, Nout / TN);
    wmma_gemm_bf16<<<gg, 128, 0, stream>>>(S.xb, S.wb, S.hlin, NN, K, Nout);

    size_t nh = (size_t)NN * Hh;
    attn_dots_k<<<nblk(nh, 256), 256, 0, stream>>>(S.hlin, a_s, a_d, S.asrc, S.adst, Hh, Cc);
    fill_k<<<nblk(nh, 256), 256, 0, stream>>>(S.mbuf, -1e30f, nh);
    fill_k<<<nblk(nh, 256), 256, 0, stream>>>(S.dbuf, 0.0f, nh);

    size_t net = (size_t)ETOT * Hh;
    edge_max_k <<<nblk(net, 256), 256, 0, stream>>>(ei, S.asrc, S.adst, S.mbuf, Hh);
    edge_exp_k <<<nblk(net, 256), 256, 0, stream>>>(ei, S.asrc, S.adst, S.mbuf, S.dbuf, S.alpha, Hh);
    edge_norm_k<<<nblk(net, 256), 256, 0, stream>>>(ei, S.dbuf, S.alpha, Hh);

    size_t nf = (size_t)NN * Nout;
    fill_k<<<nblk(nf, 256), 256, 0, stream>>>(S.hagg, 0.0f, nf);
    size_t naf = (size_t)ETOT * Nout;
    edge_aggr_k<<<nblk(naf, 256), 256, 0, stream>>>(ei, S.alpha, S.hlin, S.hagg, Hh, Cc);

    bias_act_k<<<nblk(nf, 256), 256, 0, stream>>>(S.hagg, bias, fout, Nout, do_elu);
}

extern "C" void kernel_launch(void* const* d_in, const int* in_sizes, int n_in,
                              void* d_out, int out_size, void* d_ws, size_t ws_size,
                              hipStream_t stream) {
    const float*     x     = (const float*)d_in[0];
    const long long* ei    = (const long long*)d_in[1];   // int64 in reference
    const long long* batch = (const long long*)d_in[2];   // int64 in reference
    const float* W1  = (const float*)d_in[3];
    const float* as1 = (const float*)d_in[4];
    const float* ad1 = (const float*)d_in[5];
    const float* b1  = (const float*)d_in[6];
    const float* W2  = (const float*)d_in[7];
    const float* as2 = (const float*)d_in[8];
    const float* ad2 = (const float*)d_in[9];
    const float* b2  = (const float*)d_in[10];
    const float* W3  = (const float*)d_in[11];
    const float* as3 = (const float*)d_in[12];
    const float* ad3 = (const float*)d_in[13];
    const float* b3  = (const float*)d_in[14];

    // ---- carve workspace (~84 MB) ----
    char* p = (char*)d_ws;
    auto carve = [&](size_t bytes) -> void* {
        void* r = (void*)p; p += (bytes + 255) & ~(size_t)255; return r;
    };
    Scratch S;
    S.xb    = (unsigned short*)carve((size_t)NN * 256 * 2);
    S.wb    = (unsigned int*)carve((size_t)(256 / 2) * 256 * 4);
    S.hlin  = (float*)carve((size_t)NN * 256 * 4);
    S.hagg  = (float*)carve((size_t)NN * 256 * 4);
    S.hbuf  = (float*)carve((size_t)NN * 256 * 4);
    S.asrc  = (float*)carve((size_t)NN * 4 * 4);
    S.adst  = (float*)carve((size_t)NN * 4 * 4);
    S.mbuf  = (float*)carve((size_t)NN * 4 * 4);
    S.dbuf  = (float*)carve((size_t)NN * 4 * 4);
    S.alpha = (float*)carve((size_t)ETOT * 4 * 4);
    S.psum  = (float*)carve((size_t)GG * 128 * 4);
    S.pcnt  = (float*)carve((size_t)GG * 4);

    // layer 1: 128 -> 4x64 (concat) + ELU
    run_layer(x,      128, W1, as1, ad1, b1, 4, 64, 1, S.hbuf, ei, S, stream);
    // layer 2: 256 -> 4x64 (concat) + ELU   (hbuf read only at cvt, safe to reuse)
    run_layer(S.hbuf, 256, W2, as2, ad2, b2, 4, 64, 1, S.hbuf, ei, S, stream);
    // layer 3: 256 -> 128 (heads=1, no act) -> straight into d_out h-region
    float* hout = (float*)d_out + GG * 128;
    run_layer(S.hbuf, 256, W3, as3, ad3, b3, 1, 128, 0, hout, ei, S, stream);

    // ---- global mean pool ----
    fill_k<<<nblk((size_t)GG * 128, 256), 256, 0, stream>>>(S.psum, 0.0f, (size_t)GG * 128);
    fill_k<<<1, 256, 0, stream>>>(S.pcnt, 0.0f, (size_t)GG);
    pool_sum_k<<<nblk((size_t)NN * 128, 256), 256, 0, stream>>>(batch, hout, S.psum);
    pool_cnt_k<<<nblk((size_t)NN, 256), 256, 0, stream>>>(batch, S.pcnt);
    pool_fin_k<<<nblk((size_t)GG * 128, 256), 256, 0, stream>>>(S.psum, S.pcnt, (float*)d_out);
}
